// GeometricEmbedding_23184233463967
// MI455X (gfx1250) — compile-verified, matrix operand order checked by reading, not run on hardware
//
#include <hip/hip_runtime.h>
#include <hip/hip_bf16.h>

typedef __attribute__((ext_vector_type(16))) _Float16 v16h;
typedef __attribute__((ext_vector_type(8)))  _Float16 v8h;
typedef __attribute__((ext_vector_type(8)))  float    v8f;

#define N_QUERY 131072
#define K_NBR   32
#define NB1     512          // N_QUERY / 256

// ---------------------------------------------------------------------------
// Analytic eigenvalues of a symmetric 3x3 matrix, descending order.
// ---------------------------------------------------------------------------
__device__ __forceinline__ void eig3_desc(float a00, float a01, float a02,
                                          float a11, float a12, float a22,
                                          float& e0, float& e1, float& e2) {
    float p1 = a01 * a01 + a02 * a02 + a12 * a12;
    float q  = (a00 + a11 + a22) * (1.0f / 3.0f);
    float d0 = a00 - q, d1 = a11 - q, d2 = a22 - q;
    float p2 = d0 * d0 + d1 * d1 + d2 * d2 + 2.0f * p1;
    float p  = sqrtf(p2 * (1.0f / 6.0f));
    if (p < 1e-12f) { e0 = e1 = e2 = q; return; }
    float ip  = 1.0f / p;
    float b00 = d0 * ip, b11 = d1 * ip, b22 = d2 * ip;
    float b01 = a01 * ip, b02 = a02 * ip, b12 = a12 * ip;
    float detB = b00 * (b11 * b22 - b12 * b12)
               - b01 * (b01 * b22 - b12 * b02)
               + b02 * (b01 * b12 - b11 * b02);
    float r = 0.5f * detB;
    r = fminf(1.0f, fmaxf(-1.0f, r));
    float phi  = acosf(r) * (1.0f / 3.0f);
    float twop = 2.0f * p;
    e0 = q + twop * cosf(phi);
    e2 = q + twop * cosf(phi + 2.0943951023931953f);  // + 2*pi/3
    e1 = 3.0f * q - e0 - e2;
}

// ---------------------------------------------------------------------------
// K1: per-query statistics -> feat[Q,9] f32, plus deterministic per-block
//     partial sums/sumsq (tree reduction, fixed order) for normalization.
// ---------------------------------------------------------------------------
__global__ __launch_bounds__(256)
void k_stats(const float* __restrict__ geom, const float* __restrict__ queries,
             const int* __restrict__ nbr_idx,
             float* __restrict__ feat, float* __restrict__ partials) {
    const int q  = blockIdx.x * 256 + threadIdx.x;
    const float qx = queries[q * 3 + 0];
    const float qy = queries[q * 3 + 1];
    const float qz = queries[q * 3 + 2];

    float sx = 0, sy = 0, sz = 0;
    float sxx = 0, syy = 0, szz = 0, sxy = 0, sxz = 0, syz = 0;
    float sd = 0, sd2 = 0;
    const int* ip = nbr_idx + (size_t)q * K_NBR;
#pragma unroll 4
    for (int k = 0; k < K_NBR; ++k) {
        const int idx = ip[k];
        const float* g = geom + (size_t)idx * 3;
        const float x = g[0], y = g[1], z = g[2];
        sx += x; sy += y; sz += z;
        sxx += x * x; syy += y * y; szz += z * z;
        sxy += x * y; sxz += x * z; syz += y * z;
        const float dx = x - qx, dy = y - qy, dz = z - qz;
        const float d2 = dx * dx + dy * dy + dz * dz;
        sd  += sqrtf(d2);
        sd2 += d2;
    }
    const float inv = 1.0f / (float)K_NBR;
    const float cx = sx * inv, cy = sy * inv, cz = sz * inv;
    const float Davg = sd * inv;
    const float Dvar = fmaxf(sd2 * inv - Davg * Davg, 0.0f);
    // covariance of neighbors about their centroid: E[xx^T] - c c^T
    const float c00 = sxx * inv - cx * cx;
    const float c11 = syy * inv - cy * cy;
    const float c22 = szz * inv - cz * cz;
    const float c01 = sxy * inv - cx * cy;
    const float c02 = sxz * inv - cx * cz;
    const float c12 = syz * inv - cy * cz;
    float e0, e1, e2;
    eig3_desc(c00, c01, c02, c11, c12, c22, e0, e1, e2);

    float f[9] = { (float)K_NBR, Davg, Dvar, cx - qx, cy - qy, cz - qz, e0, e1, e2 };
#pragma unroll
    for (int j = 0; j < 9; ++j) feat[(size_t)q * 9 + j] = f[j];

    // deterministic block reductions (fixed tree order)
    __shared__ float sred[256];
    const int tid = threadIdx.x;
#pragma unroll 1
    for (int j = 0; j < 9; ++j) {
        sred[tid] = f[j];
        __syncthreads();
        for (int s = 128; s > 0; s >>= 1) {
            if (tid < s) sred[tid] += sred[tid + s];
            __syncthreads();
        }
        if (tid == 0) partials[blockIdx.x * 18 + j] = sred[0];
        __syncthreads();
        sred[tid] = f[j] * f[j];
        __syncthreads();
        for (int s = 128; s > 0; s >>= 1) {
            if (tid < s) sred[tid] += sred[tid + s];
            __syncthreads();
        }
        if (tid == 0) partials[blockIdx.x * 18 + 9 + j] = sred[0];
        __syncthreads();
    }
}

// ---------------------------------------------------------------------------
// K2: final reduction -> mean / inv_std; convert weights to f16 in WMMA
//     B-operand lane order (each lane's 16 halves contiguous, 32B).
//     B-operand element map (16-bit, 32x16 K x N tile):
//       lane = hi*16 + l16; e<8  -> K = hi*8 + e
//                            e>=8 -> K = 16 + hi*8 + (e-8);  N = l16 (+ tile)
// ---------------------------------------------------------------------------
__global__ __launch_bounds__(256)
void k_reduce_prep(const float* __restrict__ partials,
                   const float* __restrict__ W1, const float* __restrict__ W2,
                   float* __restrict__ params,
                   _Float16* __restrict__ w1B, _Float16* __restrict__ w2B) {
    const int tid = threadIdx.x;
    if (tid < 9) {
        float s = 0.0f, s2 = 0.0f;
        for (int b = 0; b < NB1; ++b) {            // fixed serial order: deterministic
            s  += partials[b * 18 + tid];
            s2 += partials[b * 18 + 9 + tid];
        }
        const float Qf   = (float)N_QUERY;
        const float mean = s / Qf;
        float var = (s2 - s * s / Qf) / (Qf - 1.0f);   // unbiased (ddof=1)
        float sd  = sqrtf(fmaxf(var, 0.0f));
        if (sd < 1e-6f) sd = 1.0f;
        params[tid]     = mean;
        params[9 + tid] = 1.0f / sd;
    }
    // W1: [9,64] f32 -> padded [32,64] f16, B-operand order, 4 N-tiles
    for (int idx = tid; idx < 4 * 32 * 16; idx += 256) {
        const int e = idx & 15, lane = (idx >> 4) & 31, nt = idx >> 9;
        const int l16 = lane & 15, hi = lane >> 4;
        const int K = (e < 8) ? (hi * 8 + e) : (16 + hi * 8 + (e - 8));
        const int N = nt * 16 + l16;
        w1B[idx] = (_Float16)((K < 9) ? W1[K * 64 + N] : 0.0f);
    }
    // W2: [64,128] f32 -> f16, B-operand order, 2 k-steps x 8 N-tiles
    for (int idx = tid; idx < 2 * 8 * 32 * 16; idx += 256) {
        const int e = idx & 15, lane = (idx >> 4) & 31;
        const int nt = (idx >> 9) & 7, ks = idx >> 12;
        const int l16 = lane & 15, hi = lane >> 4;
        const int K = ks * 32 + ((e < 8) ? (hi * 8 + e) : (16 + hi * 8 + (e - 8)));
        const int N = nt * 16 + l16;
        w2B[idx] = (_Float16)W2[K * 128 + N];
    }
}

// ---------------------------------------------------------------------------
// K3: normalize features, convert to f16, pad K 9 -> 32 (A-tile row-major).
// ---------------------------------------------------------------------------
__global__ __launch_bounds__(256)
void k_norm(const float* __restrict__ feat, const float* __restrict__ params,
            _Float16* __restrict__ feat16) {
    const int q = blockIdx.x * 256 + threadIdx.x;
    _Float16 h[32];
#pragma unroll
    for (int j = 0; j < 32; ++j) h[j] = (_Float16)0.0f;
#pragma unroll
    for (int j = 0; j < 9; ++j) {
        const float v = (feat[(size_t)q * 9 + j] - params[j]) * params[9 + j];
        h[j] = (_Float16)v;
    }
    v8h* dst = (v8h*)(feat16 + (size_t)q * 32);
#pragma unroll
    for (int j = 0; j < 4; ++j) {
        v8h t;
#pragma unroll
        for (int e = 0; e < 8; ++e) t[e] = h[j * 8 + e];
        dst[j] = t;
    }
}

// ---------------------------------------------------------------------------
// K4: WMMA MLP. 8 waves/block, 16 queries/wave (128 q/block).
//     Layer1: 4x v_wmma_f32_16x16x32_f16 ; Layer2: 16x (K=64, N=128).
// ---------------------------------------------------------------------------
__global__ __launch_bounds__(256)
void k_mlp(const _Float16* __restrict__ feat16,
           const _Float16* __restrict__ w1B, const _Float16* __restrict__ w2B,
           const float* __restrict__ b1, const float* __restrict__ b2,
           float* __restrict__ out) {
    __shared__ _Float16 Hs[8][16][64];   // per-wave H staging (16 KB)

    const int tid  = threadIdx.x;
    const int w    = tid >> 5;
    const int lane = tid & 31;
    const int l16  = lane & 15;
    const int hi   = lane >> 4;
    const int q0   = blockIdx.x * 128 + w * 16;

    // ---- A operand for layer 1: row M = l16, K groups per documented layout
    const _Float16* arow = feat16 + (size_t)(q0 + l16) * 32;
    v8h alo = *(const v8h*)(arow + hi * 8);
    v8h ahi = *(const v8h*)(arow + 16 + hi * 8);
    v16h A = __builtin_shufflevector(alo, ahi, 0, 1, 2, 3, 4, 5, 6, 7,
                                     8, 9, 10, 11, 12, 13, 14, 15);

    // ---- Layer 1: feat[16x32] @ W1[32x64] -> H[16x64]
    v8f h[4];
#pragma unroll
    for (int nt = 0; nt < 4; ++nt) {
        v16h B = *(const v16h*)(w1B + ((nt * 32 + lane) << 4));
        v8f c = {};
        c = __builtin_amdgcn_wmma_f32_16x16x32_f16(false, A, false, B,
                                                   (short)0, c, false, false);
        h[nt] = c;
    }

    // bias + relu, convert to f16, stage to LDS (D layout: M = r + 8*hi)
#pragma unroll
    for (int nt = 0; nt < 4; ++nt) {
        const int N = nt * 16 + l16;
        const float bb = b1[N];
#pragma unroll
        for (int r = 0; r < 8; ++r) {
            float v = fmaxf(h[nt][r] + bb, 0.0f);
            Hs[w][r + 8 * hi][N] = (_Float16)v;
        }
    }
    __syncthreads();

    // ---- A operands for layer 2 (K=64 -> two 16x32 tiles) from LDS
    const _Float16* hrow = &Hs[w][l16][0];
    v8h a0lo = *(const v8h*)(hrow + hi * 8);
    v8h a0hi = *(const v8h*)(hrow + 16 + hi * 8);
    v8h a1lo = *(const v8h*)(hrow + 32 + hi * 8);
    v8h a1hi = *(const v8h*)(hrow + 48 + hi * 8);
    v16h A0 = __builtin_shufflevector(a0lo, a0hi, 0, 1, 2, 3, 4, 5, 6, 7,
                                      8, 9, 10, 11, 12, 13, 14, 15);
    v16h A1 = __builtin_shufflevector(a1lo, a1hi, 0, 1, 2, 3, 4, 5, 6, 7,
                                      8, 9, 10, 11, 12, 13, 14, 15);

    // ---- Layer 2: H[16x64] @ W2[64x128] -> D[16x128]
    v8f c2[8] = {};
#pragma unroll
    for (int nt = 0; nt < 8; ++nt) {
        v16h B0 = *(const v16h*)(w2B + (((0 * 8 + nt) * 32 + lane) << 4));
        c2[nt] = __builtin_amdgcn_wmma_f32_16x16x32_f16(false, A0, false, B0,
                                                        (short)0, c2[nt], false, false);
        v16h B1 = *(const v16h*)(w2B + (((1 * 8 + nt) * 32 + lane) << 4));
        c2[nt] = __builtin_amdgcn_wmma_f32_16x16x32_f16(false, A1, false, B1,
                                                        (short)0, c2[nt], false, false);
    }

    // ---- bias + relu + store (C/D layout: element (r + 8*hi, nt*16 + l16))
#pragma unroll
    for (int nt = 0; nt < 8; ++nt) {
        const int N = nt * 16 + l16;
        const float bb = b2[N];
#pragma unroll
        for (int r = 0; r < 8; ++r) {
            const float v = fmaxf(c2[nt][r] + bb, 0.0f);
            out[(size_t)(q0 + r + 8 * hi) * 128 + N] = v;
        }
    }
}

// ---------------------------------------------------------------------------
extern "C" void kernel_launch(void* const* d_in, const int* in_sizes, int n_in,
                              void* d_out, int out_size, void* d_ws, size_t ws_size,
                              hipStream_t stream) {
    const float* geom    = (const float*)d_in[0];   // [200000,3]
    const float* queries = (const float*)d_in[1];   // [131072,3]
    const int*   nbr_idx = (const int*)d_in[2];     // [131072*32]
    // d_in[3] = neighbors_row_splits (uniform: q*32) -- unused
    const float* W1 = (const float*)d_in[4];        // [9,64]
    const float* b1 = (const float*)d_in[5];        // [64]
    const float* W2 = (const float*)d_in[6];        // [64,128]
    const float* b2 = (const float*)d_in[7];        // [128]
    float* out = (float*)d_out;                     // [131072,128]

    char* ws = (char*)d_ws;
    float*    feat     = (float*)ws;                        // 4,718,592 B
    float*    partials = (float*)(ws + 4718592);            //    36,864 B
    float*    params   = (float*)(ws + 4755456);            //       256 B
    _Float16* w1B      = (_Float16*)(ws + 4755712);         //     4,096 B
    _Float16* w2B      = (_Float16*)(ws + 4759808);         //    16,384 B
    _Float16* feat16   = (_Float16*)(ws + 4776192);         // 8,388,608 B
    // total ws used: 13,164,800 B

    k_stats<<<NB1, 256, 0, stream>>>(geom, queries, nbr_idx, feat, partials);
    k_reduce_prep<<<1, 256, 0, stream>>>(partials, W1, W2, params, w1B, w2B);
    k_norm<<<NB1, 256, 0, stream>>>(feat, params, feat16);
    k_mlp<<<N_QUERY / 128, 256, 0, stream>>>(feat16, w1B, w2B, b1, b2, out);
}
